// ModulatedConv2d_63797444215155
// MI455X (gfx1250) — matrix operand order repaired and checked by reading, not compiled, not run on hardware
//
#include <hip/hip_runtime.h>
#include <cstddef>

typedef __attribute__((ext_vector_type(16))) __bf16 v16bf;
typedef __attribute__((ext_vector_type(8)))  __bf16 v8bf;
typedef __attribute__((ext_vector_type(8)))  float   v8f;
typedef __attribute__((ext_vector_type(4)))  float   v4f;
typedef __attribute__((ext_vector_type(4)))  unsigned int v4u;

#define B_SZ      16
#define IN_CH     512
#define OUT_CH    512
#define STYLE_DIM 512
#define HW        1024
#define KTAPS     9
#define KTOT      (IN_CH * KTAPS)   // 4608

__device__ __forceinline__ unsigned short f32_to_bf16(float f) {
  unsigned int u = __float_as_uint(f);
  u += 0x7FFFu + ((u >> 16) & 1u);   // round-to-nearest-even
  return (unsigned short)(u >> 16);
}
__device__ __forceinline__ unsigned int pack_bf16x2(float a, float b) {
  return (unsigned int)f32_to_bf16(a) | ((unsigned int)f32_to_bf16(b) << 16);
}

// ---------------- s[b,i] = style @ (W/sqrt(512))^T + bias + 1 ----------------
__global__ void k_affine(const float* __restrict__ style,
                         const float* __restrict__ aw,
                         const float* __restrict__ ab,
                         float* __restrict__ s) {
  int gid = blockIdx.x * blockDim.x + threadIdx.x;
  if (gid >= B_SZ * IN_CH) return;
  int b = gid >> 9, i = gid & 511;
  const float* st = style + (size_t)b * STYLE_DIM;
  const float* w  = aw + (size_t)i * STYLE_DIM;
  float acc = 0.f;
  for (int k = 0; k < STYLE_DIM; ++k) acc += st[k] * w[k];
  s[gid] = acc * 0.04419417382415922f + ab[i] + 1.0f;  // 1/sqrt(512)
}

// ---------------- q[o,i] = sum_t conv_w[o,i,t]^2 ----------------
__global__ void k_qsum(const float* __restrict__ cw, float* __restrict__ q) {
  int gid = blockIdx.x * blockDim.x + threadIdx.x;
  if (gid >= OUT_CH * IN_CH) return;
  const float* p = cw + (size_t)gid * KTAPS;
  float acc = 0.f;
#pragma unroll
  for (int t = 0; t < KTAPS; ++t) acc += p[t] * p[t];
  q[gid] = acc;
}

// ---------------- d[b,o] = rsqrt(ws^2 * sum_i s^2 q + 1e-8) ----------------
__global__ void k_demod(const float* __restrict__ s, const float* __restrict__ q,
                        float* __restrict__ d) {
  int gid = blockIdx.x * blockDim.x + threadIdx.x;
  if (gid >= B_SZ * OUT_CH) return;
  int b = gid >> 9, o = gid & 511;
  const float* sb = s + (size_t)b * IN_CH;
  const float* qo = q + (size_t)o * IN_CH;
  float acc = 0.f;
  for (int i = 0; i < IN_CH; ++i) { float si = sb[i]; acc += si * si * qo[i]; }
  d[gid] = rsqrtf(acc * (1.0f / 4608.0f) + 1e-8f);
}

// ---- wmod[b][o][t*512+i] = bf16(cw[o,i,t] * s[b,i] * ws * d[b,o]), 8 i per thread
__global__ void k_modw(const float* __restrict__ cw, const float* __restrict__ s,
                       const float* __restrict__ d, unsigned short* __restrict__ wmod) {
  size_t gid = (size_t)blockIdx.x * 256 + threadIdx.x;   // total 4718592
  int i0 = (int)((gid & 63) << 3);
  size_t r1 = gid >> 6;
  int t = (int)(r1 % 9);
  size_t r2 = r1 / 9;
  int o = (int)(r2 & 511);
  int b = (int)(r2 >> 9);
  const float* cwp = cw + ((size_t)o * IN_CH + i0) * KTAPS + t;
  const float* sp  = s + b * IN_CH + i0;
  float scale = 0.014731391274719738f * d[b * OUT_CH + o];  // 1/sqrt(4608) * demod
  v4u pkt;
#pragma unroll
  for (int j = 0; j < 4; ++j) {
    float v0 = cwp[(2 * j) * 9]     * sp[2 * j]     * scale;
    float v1 = cwp[(2 * j + 1) * 9] * sp[2 * j + 1] * scale;
    pkt[j] = pack_bf16x2(v0, v1);
  }
  *(v4u*)(wmod + (gid << 3)) = pkt;
}

// ---------------- x -> bf16, 8 per thread ----------------
__global__ void k_xcast(const float* __restrict__ x, unsigned short* __restrict__ xb) {
  size_t gid = (size_t)blockIdx.x * 256 + threadIdx.x;   // total 1048576
  size_t base = gid << 3;
  v4f x0 = *(const v4f*)(x + base);
  v4f x1 = *(const v4f*)(x + base + 4);
  v4u pkt;
  pkt[0] = pack_bf16x2(x0[0], x0[1]);
  pkt[1] = pack_bf16x2(x0[2], x0[3]);
  pkt[2] = pack_bf16x2(x1[0], x1[1]);
  pkt[3] = pack_bf16x2(x1[2], x1[3]);
  *(v4u*)(xb + base) = pkt;
}

// ---------------- implicit-GEMM conv via bf16 WMMA ----------------
// per batch: out[512,1024] = wmod[512,4608] x im2col(x)[4608,1024]
// block 128(M) x 128(N); 8 waves (4M x 2N); wave tile 32x64 = 8 WMMA accumulators.
// K loop: 16 channel-chunks (32 ch) outer x 9 taps inner. Each chunk staged once
// to LDS with a 6-row x 34-col zero-filled spatial halo, transposed to [sp][k]
// so B fragments are 2x ds_load_b128 of 8 consecutive k each. 72 WMMAs per barrier.
__launch_bounds__(256)
__global__ void k_conv_wmma(const unsigned short* __restrict__ wmod,
                            const unsigned short* __restrict__ xb,
                            float* __restrict__ out) {
  __shared__ __align__(16) unsigned short sm[2][204 * 40];  // [6*34 halo sp][32k + 8 pad]

  const int tid  = threadIdx.x;
  const int b    = blockIdx.z;
  const int m0   = blockIdx.y << 7;   // M tile (128 rows)
  const int p0   = blockIdx.x << 7;   // N tile: image rows h0..h0+3
  const int h0   = p0 >> 5;
  const int lane = tid & 31;
  const int wid  = tid >> 5;
  const int wm   = wid & 3;           // M sub (32 rows)
  const int wn   = wid >> 2;          // N sub (64 cols)
  const int laneM = lane & 15;
  const int hi    = lane >> 4;        // 0|1 half-wave

  const unsigned short* xb_b = xb + ((size_t)b << 19);  // b*512*1024
  // A row pointers for the wave's two 16-row M subtiles (ISA 16-bit A layout:
  // per lane two 8-element runs at k = 8*hi + {0..7} and 16 + 8*hi + {0..7})
  const unsigned short* wr0 =
      wmod + ((size_t)b * OUT_CH + (size_t)(m0 + wm * 32 + laneM)) * KTOT + hi * 8;
  const unsigned short* wr1 = wr0 + (size_t)16 * KTOT;

  // Packed staging descriptors: 6528 elems = 204 halo-spatial x 32 k, 26/thread.
  // pk = (ldsOff << 16) | globalOff(15b) | zeroFill(bit15); 0xFFFFFFFF = skip.
  unsigned int pk[26];
#pragma unroll
  for (int j = 0; j < 26; ++j) {
    int idx = tid + (j << 8);
    if (idx < 6528) {
      int sp = idx % 204;
      int k  = idx / 204;
      int r  = sp / 34, c = sp % 34;
      int h = h0 - 1 + r;             // image row
      int w = c - 1;                  // image col
      bool ok = ((unsigned)h < 32u) && ((unsigned)w < 32u);
      unsigned int go = ok ? (unsigned)((k << 10) + (h << 5) + w) : 0x8000u;
      pk[j] = ((unsigned)(sp * 40 + k) << 16) | go;
    } else {
      pk[j] = 0xFFFFFFFFu;
    }
  }

  v8f acc[2][4] = {};

  // stage chunk 0
#pragma unroll
  for (int j = 0; j < 26; ++j) {
    unsigned int u = pk[j];
    if (u != 0xFFFFFFFFu) {
      unsigned short v = 0;
      if (!(u & 0x8000u)) v = xb_b[u & 0x7FFFu];
      sm[0][u >> 16] = v;
    }
  }
  __syncthreads();

  for (int g = 0; g < 16; ++g) {
    const int cur = g & 1;
    if (g + 1 < 16) {
      const unsigned int cbase = (unsigned)((g + 1) << 15);  // next 32-ch chunk
#pragma unroll
      for (int j = 0; j < 26; ++j) {
        unsigned int u = pk[j];
        if (u != 0xFFFFFFFFu) {
          unsigned short v = 0;
          if (!(u & 0x8000u)) v = xb_b[cbase + (u & 0x7FFFu)];
          sm[cur ^ 1][u >> 16] = v;
        }
      }
    }

#pragma unroll
    for (int t = 0; t < 9; ++t) {
      const int dy = t / 3, dx = t % 3;       // tap offset + 1
      const int kk0 = t * 512 + (g << 5);

      v16bf a0, a1;
      *((v8bf*)&a0)     = *(const v8bf*)(wr0 + kk0);
      *((v8bf*)&a0 + 1) = *(const v8bf*)(wr0 + kk0 + 16);
      *((v8bf*)&a1)     = *(const v8bf*)(wr1 + kk0);
      *((v8bf*)&a1 + 1) = *(const v8bf*)(wr1 + kk0 + 16);

#pragma unroll
      for (int sn = 0; sn < 4; ++sn) {
        // column n = wn*64 + sn*16 + laneM -> halo row wn*2+(sn>>1)+dy,
        // halo col (sn&1)*16 + laneM + dx; k = 8*hi + {0..7} and 16 + 8*hi + {0..7} per lane.
        const unsigned short* bb =
            &sm[cur][(((wn << 1) + (sn >> 1) + dy) * 34 + ((sn & 1) << 4) + laneM + dx) * 40
                     + (hi << 3)];
        v16bf bf;
        *((v8bf*)&bf)     = *(const v8bf*)(bb);
        *((v8bf*)&bf + 1) = *(const v8bf*)(bb + 16);

        acc[0][sn] = __builtin_amdgcn_wmma_f32_16x16x32_bf16(false, a0, false, bf, (short)0, acc[0][sn], false, false);
        acc[1][sn] = __builtin_amdgcn_wmma_f32_16x16x32_bf16(false, a1, false, bf, (short)0, acc[1][sn], false, false);
      }
    }
    __syncthreads();
  }

  // epilogue: C/D layout m = r + 8*hi, n = lane%16
  const int mbase = m0 + wm * 32 + hi * 8;
  const int nbase = p0 + wn * 64 + laneM;
  float* ob = out + (((size_t)b * OUT_CH + mbase) << 10) + nbase;
#pragma unroll
  for (int smi = 0; smi < 2; ++smi) {
#pragma unroll
    for (int sn = 0; sn < 4; ++sn) {
#pragma unroll
      for (int r = 0; r < 8; ++r) {
        ob[(((size_t)(smi * 16 + r)) << 10) + sn * 16] = acc[smi][sn][r];
      }
    }
  }
}

extern "C" void kernel_launch(void* const* d_in, const int* in_sizes, int n_in,
                              void* d_out, int out_size, void* d_ws, size_t ws_size,
                              hipStream_t stream) {
  (void)in_sizes; (void)n_in; (void)out_size; (void)ws_size;
  const float* x     = (const float*)d_in[0];
  const float* style = (const float*)d_in[1];
  const float* aw    = (const float*)d_in[2];
  const float* ab    = (const float*)d_in[3];
  const float* cw    = (const float*)d_in[4];
  float* out = (float*)d_out;

  char* ws = (char*)d_ws;
  float* s          = (float*)(ws);                                  //  32 KB
  float* d          = (float*)(ws + 32768);                          //  32 KB
  float* q          = (float*)(ws + 65536);                          //   1 MB
  unsigned short* xb   = (unsigned short*)(ws + 65536 + 1048576);    //  16 MB
  unsigned short* wmod = (unsigned short*)(ws + 65536 + 1048576 + 16777216); // 75.5 MB

  k_affine<<<(B_SZ * IN_CH) / 256, 256, 0, stream>>>(style, aw, ab, s);
  k_qsum<<<(OUT_CH * IN_CH) / 256, 256, 0, stream>>>(cw, q);
  k_demod<<<(B_SZ * OUT_CH) / 256, 256, 0, stream>>>(s, q, d);
  k_modw<<<(unsigned)(((size_t)B_SZ * OUT_CH * KTOT) / 8 / 256), 256, 0, stream>>>(cw, s, d, wmod);
  k_xcast<<<(unsigned)(((size_t)B_SZ * IN_CH * HW) / 8 / 256), 256, 0, stream>>>(x, xb);

  dim3 grid(HW / 128, OUT_CH / 128, B_SZ);
  k_conv_wmma<<<grid, 256, 0, stream>>>(wmod, xb, out);
}
